// RandomizedBertSelfAttention_45509473468722
// MI455X (gfx1250) — matrix operand order repaired and checked by reading, not compiled, not run on hardware
//
#include <hip/hip_runtime.h>
#include <hip/hip_bf16.h>

#define SEQ    4096
#define HIDDEN 768
#define NHEAD  12
#define DHEAD  64
#define NKT    (SEQ / 64)

typedef __attribute__((ext_vector_type(16))) _Float16 v16h;
typedef __attribute__((ext_vector_type(8)))  _Float16 v8h_t;
typedef __attribute__((ext_vector_type(8)))  float    v8f;
typedef __attribute__((ext_vector_type(4)))  int      v4i;

// --- CDNA5 async global->LDS copy (ASYNCcnt path), with synchronous fallback ---
#if defined(__has_builtin)
#if __has_builtin(__builtin_amdgcn_global_load_async_to_lds_b128) && \
    __has_builtin(__builtin_amdgcn_s_wait_asynccnt)
#define USE_ASYNC_COPY 1
#endif
#endif
#ifndef USE_ASYNC_COPY
#define USE_ASYNC_COPY 0
#endif

typedef __attribute__((address_space(1))) v4i as1_v4i;
typedef __attribute__((address_space(3))) v4i as3_v4i;

__device__ __forceinline__ void async_cp16(const _Float16* g, _Float16* l) {
#if USE_ASYNC_COPY
  __builtin_amdgcn_global_load_async_to_lds_b128(
      (as1_v4i*)g, (as3_v4i*)l, 0, 0);
#else
  *(uint4*)l = *(const uint4*)g;
#endif
}

// Wait until at most the newest tile-group (4 async instrs/wave) is outstanding.
__device__ __forceinline__ void async_wait_keep_one() {
#if USE_ASYNC_COPY
  __builtin_amdgcn_s_wait_asynccnt(4);
#endif
}
__device__ __forceinline__ void async_wait_all() {
#if USE_ASYNC_COPY
  __builtin_amdgcn_s_wait_asynccnt(0);
#endif
}

union FragU { v16h v; v8h_t h[2]; };

// A-matrix fragment (16x32 f16), LDS row-major [m][K], ld = row stride in halfs.
// Layout (ISA 7.12.2): lanes 0-15 -> M=lane, VGPR0-3 hold K = g*8 + 0..7,
// VGPR4-7 hold K = 16 + g*8 + 0..7  (g = lane/16).
__device__ __forceinline__ v16h lds_frag_a(const _Float16* p0, int ld, int lane) {
  const int m = lane & 15, g = lane >> 4;
  const _Float16* p = p0 + m * ld + g * 8;
  FragU u;
  u.h[0] = *(const v8h_t*)(p);
  u.h[1] = *(const v8h_t*)(p + 16);
  return u.v;
}

// B-matrix fragment (32x16 f16), LDS row-major [n][K]:
// lane n = lane%16 is the N column, K = g*8 + 0..7 then 16 + g*8 + 0..7 (g = lane/16).
__device__ __forceinline__ v16h lds_frag_b(const _Float16* p0, int ld, int lane) {
  const int n = lane & 15, g = lane >> 4;
  const _Float16* p = p0 + n * ld + g * 8;
  FragU u;
  u.h[0] = *(const v8h_t*)(p);
  u.h[1] = *(const v8h_t*)(p + 16);
  return u.v;
}

// out[s, o] = sum_d x[s,d]*W[o,d] + b[o], f16 output.
// transposed==0: out laid out [head][s][d]; transposed!=0: out laid out [head][d][s]
// (V is stored transposed so the attention kernel's P*V B-fragments are K-contiguous
//  and its tiles can be staged with async 16B copies, no LDS transpose needed).
__global__ __launch_bounds__(256) void qkv_proj_kernel(
    const float* __restrict__ x, const float* __restrict__ W,
    const float* __restrict__ bias, _Float16* __restrict__ out, int transposed)
{
  __shared__ __align__(16) _Float16 As[128 * 32];
  __shared__ __align__(16) _Float16 Bs[64 * 32];

  const int tid  = threadIdx.x;
  const int lane = tid & 31;
  const int w    = tid >> 5;
  const int wm   = w & 3;            // M-tile pair index
  const int wn   = w >> 2;           // N-tile pair index
  const int s0   = blockIdx.x * 128;
  const int n0   = blockIdx.y * 64;  // == head * 64

  v8f acc[2][2] = {};

  for (int kc = 0; kc < HIDDEN; kc += 32) {
    // Stage A tile 128x32 (f32 -> f16), 4 float4 per thread.
    #pragma unroll
    for (int i = 0; i < 4; ++i) {
      int idx = tid + i * 256;             // 0..1023, each covers 4 elems
      int row = idx >> 3;
      int c4  = (idx & 7) * 4;
      const float4 f = *(const float4*)(x + (size_t)(s0 + row) * HIDDEN + kc + c4);
      _Float16* dst = As + row * 32 + c4;
      dst[0] = (_Float16)f.x; dst[1] = (_Float16)f.y;
      dst[2] = (_Float16)f.z; dst[3] = (_Float16)f.w;
    }
    // Stage B tile 64x32 (rows of W are K-contiguous).
    #pragma unroll
    for (int i = 0; i < 2; ++i) {
      int idx = tid + i * 256;             // 0..511
      int row = idx >> 3;
      int c4  = (idx & 7) * 4;
      const float4 f = *(const float4*)(W + (size_t)(n0 + row) * HIDDEN + kc + c4);
      _Float16* dst = Bs + row * 32 + c4;
      dst[0] = (_Float16)f.x; dst[1] = (_Float16)f.y;
      dst[2] = (_Float16)f.z; dst[3] = (_Float16)f.w;
    }
    __syncthreads();

    v16h a0 = lds_frag_a(As + (wm * 2 + 0) * 16 * 32, 32, lane);
    v16h a1 = lds_frag_a(As + (wm * 2 + 1) * 16 * 32, 32, lane);
    v16h b0 = lds_frag_b(Bs + (wn * 2 + 0) * 16 * 32, 32, lane);
    v16h b1 = lds_frag_b(Bs + (wn * 2 + 1) * 16 * 32, 32, lane);
    acc[0][0] = __builtin_amdgcn_wmma_f32_16x16x32_f16(false, a0, false, b0, (short)0, acc[0][0], false, false);
    acc[0][1] = __builtin_amdgcn_wmma_f32_16x16x32_f16(false, a0, false, b1, (short)0, acc[0][1], false, false);
    acc[1][0] = __builtin_amdgcn_wmma_f32_16x16x32_f16(false, a1, false, b0, (short)0, acc[1][0], false, false);
    acc[1][1] = __builtin_amdgcn_wmma_f32_16x16x32_f16(false, a1, false, b1, (short)0, acc[1][1], false, false);
    __syncthreads();
  }

  // C/D layout: lane n = lane%16 is N, VGPR r is M = (lane/16)*8 + r within tile.
  const int g = lane >> 4, n = lane & 15;
  #pragma unroll
  for (int j = 0; j < 2; ++j) {
    const int col = (wn * 2 + j) * 16 + n;   // 0..63 == d within head
    const float bv = bias[n0 + col];
    #pragma unroll
    for (int i = 0; i < 2; ++i) {
      const int mrow = s0 + (wm * 2 + i) * 16 + g * 8;
      if (!transposed) {
        _Float16* op = out + (size_t)blockIdx.y * SEQ * DHEAD + (size_t)mrow * DHEAD + col;
        #pragma unroll
        for (int r = 0; r < 8; ++r)
          op[(size_t)r * DHEAD] = (_Float16)(acc[i][j][r] + bv);
      } else {
        _Float16* op = out + (size_t)blockIdx.y * DHEAD * SEQ + (size_t)col * SEQ + mrow;
        #pragma unroll
        for (int r = 0; r < 8; ++r)
          op[r] = (_Float16)(acc[i][j][r] + bv);
      }
    }
  }
}

// Flash attention: block = 64 queries x 1 head; 8 waves; online softmax.
// Q,K: [head][s][d] f16.  V: [head][d][s] f16 (pre-transposed).
// K/V tiles are double-buffered and prefetched with async copies.
__global__ __launch_bounds__(256) void attn_kernel(
    const _Float16* __restrict__ Q, const _Float16* __restrict__ K,
    const _Float16* __restrict__ V, float* __restrict__ out)
{
  __shared__ __align__(16) _Float16 Qs[64 * 64];
  __shared__ __align__(16) _Float16 Ks[2][64 * 64];
  __shared__ __align__(16) _Float16 Vt[2][64 * 64];   // [d][key]
  __shared__ __align__(16) float    Sb[64 * 64];      // f16 P tile aliased into row fronts
  __shared__ float rowm[64], rowl[64], rowf[64];

  const int tid  = threadIdx.x, lane = tid & 31, w = tid >> 5;
  const int wm   = w & 3;      // query 16-row tile
  const int wn   = w >> 2;     // key/dh 32-col half
  const int qt   = blockIdx.x, head = blockIdx.y;
  const size_t hbase = (size_t)head * SEQ * DHEAD;
  const float scale = 0.125f;  // 1/sqrt(64)

  _Float16* Pb = (_Float16*)Sb;   // P row r occupies bytes [r*256, r*256+128)

  const int crow = tid >> 3, cc8 = (tid & 7) * 8;   // 16B-chunk coords, +32 rows on i=1

  // Q tile + first K/V tile: async 16B chunks.
  #pragma unroll
  for (int i = 0; i < 2; ++i) {
    int row = crow + i * 32;
    async_cp16(Q + hbase + (size_t)(qt * 64 + row) * 64 + cc8, Qs + row * 64 + cc8);
  }
  #pragma unroll
  for (int i = 0; i < 2; ++i) {
    int row = crow + i * 32;
    async_cp16(K + hbase + (size_t)row * 64 + cc8,       Ks[0] + row * 64 + cc8);
    async_cp16(V + hbase + (size_t)row * SEQ + cc8,      Vt[0] + row * 64 + cc8);
  }
  if (tid < 64) { rowm[tid] = -1e30f; rowl[tid] = 0.0f; }

  v8f o0 = {}, o1 = {};

  for (int kt = 0; kt < NKT; ++kt) {
    const int cur = kt & 1;
    // Prefetch next tile into the other buffer (its readers finished last iter).
    if (kt + 1 < NKT) {
      const int nb = kt + 1, nxt = cur ^ 1;
      #pragma unroll
      for (int i = 0; i < 2; ++i) {
        int row = crow + i * 32;
        async_cp16(K + hbase + (size_t)(nb * 64 + row) * 64 + cc8, Ks[nxt] + row * 64 + cc8);
        async_cp16(V + hbase + (size_t)row * SEQ + nb * 64 + cc8,  Vt[nxt] + row * 64 + cc8);
      }
      async_wait_keep_one();   // tile kt (and Q) landed; tile kt+1 still in flight
    } else {
      async_wait_all();
    }
    __syncthreads();

    // S = Q * K^T (each wave: 16 q-rows x 32 k-cols)
    v8f s0 = {}, s1 = {};
    #pragma unroll
    for (int kc = 0; kc < 2; ++kc) {
      v16h a  = lds_frag_a(Qs + wm * 16 * 64 + kc * 32, 64, lane);
      v16h b0 = lds_frag_b(Ks[cur] + (wn * 2 + 0) * 16 * 64 + kc * 32, 64, lane);
      v16h b1 = lds_frag_b(Ks[cur] + (wn * 2 + 1) * 16 * 64 + kc * 32, 64, lane);
      s0 = __builtin_amdgcn_wmma_f32_16x16x32_f16(false, a, false, b0, (short)0, s0, false, false);
      s1 = __builtin_amdgcn_wmma_f32_16x16x32_f16(false, a, false, b1, (short)0, s1, false, false);
    }
    {
      const int g = lane >> 4, n = lane & 15;
      #pragma unroll
      for (int r = 0; r < 8; ++r) {
        int row = wm * 16 + g * 8 + r;
        Sb[row * 64 + (wn * 2 + 0) * 16 + n] = s0[r] * scale;
        Sb[row * 64 + (wn * 2 + 1) * 16 + n] = s1[r] * scale;
      }
    }
    __syncthreads();

    // Online softmax: 4 consecutive lanes per row, 16 columns each, held in regs.
    {
      const int sr = tid >> 2;          // row
      const int c0 = (tid & 3) * 16;    // column base
      float v[16];
      #pragma unroll
      for (int j = 0; j < 16; ++j) v[j] = Sb[sr * 64 + c0 + j];
      const float m_old = rowm[sr];
      float tmax = m_old;
      #pragma unroll
      for (int j = 0; j < 16; ++j) tmax = fmaxf(tmax, v[j]);
      tmax = fmaxf(tmax, __shfl_xor(tmax, 1, 32));
      tmax = fmaxf(tmax, __shfl_xor(tmax, 2, 32));
      float sum = 0.0f;
      _Float16* prow = Pb + sr * 128 + c0;   // alias: stays within own Sb row
      #pragma unroll
      for (int j = 0; j < 16; ++j) {
        float p = __expf(v[j] - tmax);
        prow[j] = (_Float16)p;
        sum += p;
      }
      sum += __shfl_xor(sum, 1, 32);
      sum += __shfl_xor(sum, 2, 32);
      if ((tid & 3) == 0) {
        float f = __expf(m_old - tmax);
        rowl[sr] = rowl[sr] * f + sum;
        rowm[sr] = tmax;
        rowf[sr] = f;
      }
    }
    __syncthreads();

    // Rescale running context, then ctx += P * V (wave: 16 q-rows x 32 dh-cols).
    {
      const int g = lane >> 4;
      #pragma unroll
      for (int r = 0; r < 8; ++r) {
        float f = rowf[wm * 16 + g * 8 + r];
        o0[r] *= f; o1[r] *= f;
      }
    }
    #pragma unroll
    for (int kc = 0; kc < 2; ++kc) {
      v16h a  = lds_frag_a(Pb + wm * 16 * 128 + kc * 32, 128, lane);
      v16h b0 = lds_frag_b(Vt[cur] + (wn * 2 + 0) * 16 * 64 + kc * 32, 64, lane);
      v16h b1 = lds_frag_b(Vt[cur] + (wn * 2 + 1) * 16 * 64 + kc * 32, 64, lane);
      o0 = __builtin_amdgcn_wmma_f32_16x16x32_f16(false, a, false, b0, (short)0, o0, false, false);
      o1 = __builtin_amdgcn_wmma_f32_16x16x32_f16(false, a, false, b1, (short)0, o1, false, false);
    }
    __syncthreads();   // publishes completion of reads of Ks/Vt[cur], Sb/Pb
  }

  // Normalize and store: out[s][head*64 + d], fp32.
  const int g = lane >> 4, n = lane & 15;
  #pragma unroll
  for (int r = 0; r < 8; ++r) {
    int row = wm * 16 + g * 8 + r;
    float linv = 1.0f / rowl[row];
    size_t s = (size_t)(qt * 64 + row);
    out[s * HIDDEN + head * 64 + (wn * 2 + 0) * 16 + n] = o0[r] * linv;
    out[s * HIDDEN + head * 64 + (wn * 2 + 1) * 16 + n] = o1[r] * linv;
  }
}

extern "C" void kernel_launch(void* const* d_in, const int* in_sizes, int n_in,
                              void* d_out, int out_size, void* d_ws, size_t ws_size,
                              hipStream_t stream) {
  const float* x  = (const float*)d_in[0];
  const float* Wq = (const float*)d_in[1];
  const float* bq = (const float*)d_in[2];
  const float* Wk = (const float*)d_in[3];
  const float* bk = (const float*)d_in[4];
  const float* Wv = (const float*)d_in[5];
  const float* bv = (const float*)d_in[6];
  float* out = (float*)d_out;

  const size_t perbuf = (size_t)NHEAD * SEQ * DHEAD;   // f16 elements
  _Float16* qb = (_Float16*)d_ws;
  _Float16* kb = qb + perbuf;
  _Float16* vb = kb + perbuf;

  dim3 g1(SEQ / 128, HIDDEN / 64);
  qkv_proj_kernel<<<g1, 256, 0, stream>>>(x, Wq, bq, qb, 0);
  qkv_proj_kernel<<<g1, 256, 0, stream>>>(x, Wk, bk, kb, 0);
  qkv_proj_kernel<<<g1, 256, 0, stream>>>(x, Wv, bv, vb, 1);

  dim3 g2(SEQ / 64, NHEAD);
  attn_kernel<<<g2, 256, 0, stream>>>(qb, kb, vb, out);
}
